// MultiHeadAttentionModule_25761213841936
// MI455X (gfx1250) — matrix operand order repaired and checked by reading, not compiled, not run on hardware
//
#include <hip/hip_runtime.h>

// ---------------- problem constants ----------------
#define B_    2
#define S_    2048
#define HID   1024
#define NH    16
#define HDIM  64
#define NQ    3072            // 3*HID
#define MT    4096            // B_*S_
#define NCH   16              // S_/128 chunks for the prefix scan

// ---------------- GEMM tiling ----------------
#define BM 128
#define BN 128
#define BK 32
#define KP 40                 // LDS row pitch in bf16 elems (32 + 8 pad -> 80B, conflict free)

typedef __attribute__((ext_vector_type(16))) __bf16        v16bf;
typedef __attribute__((ext_vector_type(8)))  float         v8f;
typedef __attribute__((ext_vector_type(4)))  unsigned int  u32x4;
typedef __attribute__((ext_vector_type(2)))  unsigned int  u32x2;
typedef __attribute__((ext_vector_type(4)))  float         f32x4;

// ---------------- fp32 -> (hi,lo) bf16 split ----------------
__device__ __forceinline__ unsigned short bf16_rne(float f) {
    unsigned u = __float_as_uint(f);
    u += 0x7FFFu + ((u >> 16) & 1u);           // round-to-nearest-even
    return (unsigned short)(u >> 16);
}
__device__ __forceinline__ float bf16_to_f(unsigned short h) {
    return __uint_as_float(((unsigned)h) << 16);
}
__device__ __forceinline__ unsigned pack2(unsigned short a, unsigned short b) {
    return (unsigned)a | ((unsigned)b << 16);
}

// A-operand fragment (16x32, MxK): lanes 0-15 hold M=lane,
// K=0..15; lanes 16-31 hold M=lane-16, K=16..31 (matches B fragment).
__device__ __forceinline__ v16bf load_fragA(const unsigned short* lds, int row0, int lane) {
    const unsigned short* p = lds + (row0 + (lane & 15)) * KP + ((lane >> 4) << 4);
    union { u32x4 q[2]; v16bf v; } f;
    f.q[0] = *(const u32x4*)(p);        // K 0..7   (or 16..23)
    f.q[1] = *(const u32x4*)(p + 8);    // K 8..15  (or 24..31)
    return f.v;
}
// B-operand fragment (32x16, KxN), LDS holds B transposed as [N][K]: same k order as A,
// lanes 0-15 hold N=lane, K=0..15; lanes 16-31 hold
// N=lane-16, K=16..31 (contiguous 16-value K-runs).
__device__ __forceinline__ v16bf load_fragB(const unsigned short* lds, int row0, int lane) {
    const unsigned short* p = lds + (row0 + (lane & 15)) * KP + ((lane >> 4) << 4);
    union { u32x4 q[2]; v16bf v; } f;
    f.q[0] = *(const u32x4*)(p);        // K 0..7   (or 16..23)
    f.q[1] = *(const u32x4*)(p + 8);    // K 8..15  (or 24..31)
    return f.v;
}

// ================= K1: qkv = x @ wqkv via bf16x3 split WMMA =================
__global__ __launch_bounds__(256)
void qkv_gemm(const float* __restrict__ X, const float* __restrict__ W,
              float* __restrict__ QKV) {
    __shared__ __align__(16) unsigned short Ah[BM * KP];
    __shared__ __align__(16) unsigned short Al[BM * KP];
    __shared__ __align__(16) unsigned short Bh[BN * KP];
    __shared__ __align__(16) unsigned short Bl[BN * KP];

    const int t    = threadIdx.x;
    const int lane = t & 31;
    const int wave = t >> 5;
    const int wm   = (wave >> 2) * 64;   // wave row offset inside block tile
    const int wn   = (wave & 3) * 32;    // wave col offset inside block tile
    const int m0   = blockIdx.x * BM;
    const int n0   = blockIdx.y * BN;

    v8f acc[4][2];
    const v8f vzero = {0.f, 0.f, 0.f, 0.f, 0.f, 0.f, 0.f, 0.f};
#pragma unroll
    for (int i = 0; i < 4; ++i)
#pragma unroll
        for (int j = 0; j < 2; ++j) acc[i][j] = vzero;

#pragma unroll 1
    for (int k0 = 0; k0 < HID; k0 += BK) {
        __syncthreads();   // previous iteration's frag reads done

        // ---- stage A tile (128 x 32 f32) as split bf16, layout [M][K] ----
        {
            const int r  = t >> 3;          // 0..31
            const int c4 = (t & 7) << 2;    // 0,4,...,28
#pragma unroll
            for (int p = 0; p < 4; ++p) {
                const int row = p * 32 + r;
                f32x4 v = *(const f32x4*)(X + (size_t)(m0 + row) * HID + k0 + c4);
                unsigned short h[4], l[4];
#pragma unroll
                for (int e = 0; e < 4; ++e) {
                    h[e] = bf16_rne(v[e]);
                    l[e] = bf16_rne(v[e] - bf16_to_f(h[e]));
                }
                u32x2 hp = {pack2(h[0], h[1]), pack2(h[2], h[3])};
                u32x2 lp = {pack2(l[0], l[1]), pack2(l[2], l[3])};
                *(u32x2*)(Ah + row * KP + c4) = hp;
                *(u32x2*)(Al + row * KP + c4) = lp;
            }
        }
        // ---- stage B tile (32 x 128 f32) transposed as [N][K] split bf16 ----
        {
            const int n  = t & 127;
            const int kg = t >> 7;          // 0 or 1 (K half)
            const float* src = W + (size_t)(k0 + kg * 16) * NQ + n0 + n;
            unsigned short hh[16], ll[16];
#pragma unroll
            for (int i = 0; i < 16; ++i) {
                float v = src[(size_t)i * NQ];
                hh[i] = bf16_rne(v);
                ll[i] = bf16_rne(v - bf16_to_f(hh[i]));
            }
            u32x4 hq0 = {pack2(hh[0], hh[1]),  pack2(hh[2], hh[3]),
                         pack2(hh[4], hh[5]),  pack2(hh[6], hh[7])};
            u32x4 hq1 = {pack2(hh[8], hh[9]),  pack2(hh[10], hh[11]),
                         pack2(hh[12], hh[13]), pack2(hh[14], hh[15])};
            u32x4 lq0 = {pack2(ll[0], ll[1]),  pack2(ll[2], ll[3]),
                         pack2(ll[4], ll[5]),  pack2(ll[6], ll[7])};
            u32x4 lq1 = {pack2(ll[8], ll[9]),  pack2(ll[10], ll[11]),
                         pack2(ll[12], ll[13]), pack2(ll[14], ll[15])};
            u32x4* dh = (u32x4*)(Bh + n * KP + (kg << 4));
            u32x4* dl = (u32x4*)(Bl + n * KP + (kg << 4));
            dh[0] = hq0; dh[1] = hq1;
            dl[0] = lq0; dl[1] = lq1;
        }
        __syncthreads();

        // ---- WMMA: hi*hi + hi*lo + lo*hi (split fp32 emulation) ----
        v16bf ah[4], al[4], bh[2], bl[2];
#pragma unroll
        for (int i = 0; i < 4; ++i) {
            ah[i] = load_fragA(Ah, wm + i * 16, lane);
            al[i] = load_fragA(Al, wm + i * 16, lane);
        }
#pragma unroll
        for (int j = 0; j < 2; ++j) {
            bh[j] = load_fragB(Bh, wn + j * 16, lane);
            bl[j] = load_fragB(Bl, wn + j * 16, lane);
        }
#pragma unroll
        for (int i = 0; i < 4; ++i) {
#pragma unroll
            for (int j = 0; j < 2; ++j) {
                acc[i][j] = __builtin_amdgcn_wmma_f32_16x16x32_bf16(
                    false, al[i], false, bh[j], (short)0, acc[i][j], false, false);
                acc[i][j] = __builtin_amdgcn_wmma_f32_16x16x32_bf16(
                    false, ah[i], false, bl[j], (short)0, acc[i][j], false, false);
                acc[i][j] = __builtin_amdgcn_wmma_f32_16x16x32_bf16(
                    false, ah[i], false, bh[j], (short)0, acc[i][j], false, false);
            }
        }
    }

    // ---- epilogue: C/D layout VGPR r -> M=r (lanes 0-15) / M=r+8 (16-31) ----
    const int mhalf = (lane >> 4) * 8;
    const int cn    = lane & 15;
#pragma unroll
    for (int i = 0; i < 4; ++i) {
#pragma unroll
        for (int j = 0; j < 2; ++j) {
            const int col = n0 + wn + j * 16 + cn;
#pragma unroll
            for (int r = 0; r < 8; ++r) {
                const int row = m0 + wm + i * 16 + mhalf + r;
                QKV[(size_t)row * NQ + col] = acc[i][j][r];
            }
        }
    }
}

// ======== K2: wexp[b,s,h] = exp(dot(q_s, k_s)/8)  (diagonal scores) ========
__global__ __launch_bounds__(256)
void diag_exp(const float* __restrict__ QKV, float* __restrict__ WEXP) {
    const int row = blockIdx.x;          // b*S + s
    const int t   = threadIdx.x;
    const int h   = t >> 4;              // 0..15
    const int d   = (t & 15) << 2;       // 0,4,...,60
    const float* base = QKV + (size_t)row * NQ + h * 192 + d;
    f32x4 q = *(const f32x4*)(base);
    f32x4 k = *(const f32x4*)(base + 64);
    float p = q[0] * k[0] + q[1] * k[1] + q[2] * k[2] + q[3] * k[3];
    p += __shfl_xor(p, 8, 16);
    p += __shfl_xor(p, 4, 16);
    p += __shfl_xor(p, 2, 16);
    p += __shfl_xor(p, 1, 16);
    if ((t & 15) == 0) WEXP[(size_t)row * NH + h] = expf(p * 0.125f);
}

// ======== K3: per-chunk sums of V along s (for the prefix scan) ========
__global__ __launch_bounds__(64)
void chunk_sums(const float* __restrict__ QKV, float* __restrict__ PART) {
    const int blk = blockIdx.x;          // (b*NH + h)*NCH + c
    const int c   = blk & (NCH - 1);
    const int bh  = blk >> 4;            // b*NH + h
    const int b   = bh >> 4;
    const int h   = bh & 15;
    const int d   = threadIdx.x;         // 0..63
    size_t base = ((size_t)b * S_ + (size_t)c * 128) * NQ + h * 192 + 128 + d;
    float s = 0.f;
#pragma unroll 4
    for (int i = 0; i < 128; ++i) s += QKV[base + (size_t)i * NQ];
    PART[(size_t)blk * HDIM + d] = s;
}

// ======== K4: serial scan within chunk + softmax epilogue + transpose ========
// out[b,s,h*64+d] = (P_s + w_s * v_s) / (s + w_s),  P_s = sum_{j<s} v_j
__global__ __launch_bounds__(64)
void scan_out(const float* __restrict__ QKV, const float* __restrict__ WEXP,
              const float* __restrict__ PART, float* __restrict__ OUT) {
    const int blk = blockIdx.x;
    const int c   = blk & (NCH - 1);
    const int bh  = blk >> 4;
    const int b   = bh >> 4;
    const int h   = bh & 15;
    const int d   = threadIdx.x;

    float P = 0.f;
    for (int cc = 0; cc < c; ++cc)
        P += PART[((size_t)bh * NCH + cc) * HDIM + d];

    const int s0 = c * 128;
    const size_t rowbase = (size_t)b * S_ + s0;
#pragma unroll 1
    for (int i = 0; i < 128; ++i) {
        const size_t row = rowbase + i;
        const float v = QKV[row * NQ + h * 192 + 128 + d];
        const float w = WEXP[row * NH + h];
        const float denom = (float)(s0 + i) + w;
        OUT[row * HID + h * HDIM + d] = (P + w * v) / denom;
        P += v;
    }
}

// =============================== launcher ===============================
extern "C" void kernel_launch(void* const* d_in, const int* in_sizes, int n_in,
                              void* d_out, int out_size, void* d_ws, size_t ws_size,
                              hipStream_t stream) {
    const float* x = (const float*)d_in[0];   // [B,S,HID]
    const float* w = (const float*)d_in[1];   // [HID, 3*HID]
    float* out  = (float*)d_out;              // [B,S,HID] f32
    float* qkv  = (float*)d_ws;               // MT*NQ floats (50.3 MB)
    float* wexp = qkv + (size_t)MT * NQ;      // MT*NH floats
    float* part = wexp + (size_t)MT * NH;     // B_*NH*NCH*HDIM floats

    dim3 grid(MT / BM, NQ / BN);              // 32 x 24
    qkv_gemm<<<grid, 256, 0, stream>>>(x, w, qkv);
    diag_exp<<<MT, 256, 0, stream>>>(qkv, wexp);
    chunk_sums<<<B_ * NH * NCH, 64, 0, stream>>>(qkv, part);
    scan_out<<<B_ * NH * NCH, 64, 0, stream>>>(qkv, wexp, part, out);
}
